// Memory_30571577213131
// MI455X (gfx1250) — matrix operand order repaired and checked by reading, not compiled, not run on hardware
//
#include <hip/hip_runtime.h>
#include <hip/hip_bf16.h>
#include <cstdint>
#include <cmath>

// ---------------------------------------------------------------------------
// Types / helpers
// ---------------------------------------------------------------------------
typedef __attribute__((ext_vector_type(16))) __bf16 v16bf;
typedef __attribute__((ext_vector_type(8)))  float  v8f;

__device__ __forceinline__ unsigned short f2bf(float f) {
  unsigned u = __float_as_uint(f);
  u += 0x7FFFu + ((u >> 16) & 1u);          // round-to-nearest-even
  return (unsigned short)(u >> 16);
}

union FragBF { uint4 u[2]; v16bf v; };

constexpr int KSTEP = 32, LDSS = KSTEP + 8;   // padded LDS row (halves)

// CDNA5 async copy: global -> LDS, 16B per lane, tracked by ASYNCcnt.
__device__ __forceinline__ void async_load_b128(unsigned lds_off, const void* gptr) {
  asm volatile("global_load_async_to_lds_b128 %0, %1, off"
               :: "v"(lds_off), "v"(gptr) : "memory");
}
__device__ __forceinline__ void wait_asynccnt0() {
  asm volatile("s_wait_asynccnt 0x0" ::: "memory");
}

// ---------------------------------------------------------------------------
// Tiled WMMA GEMM:  C[M,N] = alpha * A[M,K] * B[N,K]^T + bias[N]
//   Block tile BMxBN (8 waves, each computing a 64x64 sub-tile via 16 WMMAs),
//   K staged in 32-wide steps through double-buffered LDS filled with
//   GLOBAL_LOAD_ASYNC_TO_LDS_B128 (ASYNCcnt pipeline).
//   omode: 0 -> f32 C[m*ldc+n], 1 -> bf16 C[m*ldc+n], 2 -> bf16 C[n*ldc+m]
//   batched via blockIdx.z with element strides sA/sB/sC (sB=0 => shared B)
// Requires M%BM==0, N%BN==0, K%32==0 (true for all GEMMs in this model).
// ---------------------------------------------------------------------------
template <int BM, int BN>
__global__ __launch_bounds__(256) void gemm_bf16_kernel(
    const unsigned short* __restrict__ A, long long sA,
    const unsigned short* __restrict__ Bm, long long sB,
    void* __restrict__ Cv, long long sC,
    const float* __restrict__ bias,
    int K, int lda, int ldb, int ldc,
    float alpha, int omode)
{
  static_assert(BM + BN == 384, "chunk math assumes BM+BN==384");
  constexpr int WM_WAVES = BM / 64;
  constexpr int ROWS = BM + BN;
  __shared__ __align__(16) unsigned short ldsbuf[2][ROWS * LDSS];

  const int bz = blockIdx.z;
  const unsigned short* Ab = A + (long long)bz * sA + (long long)blockIdx.y * BM * lda;
  const unsigned short* Bb = Bm + (long long)bz * sB + (long long)blockIdx.x * BN * ldb;

  const int t    = threadIdx.x;
  const int lane = t & 31;
  const int wave = t >> 5;
  const int wm   = (wave % WM_WAVES) * 64;
  const int wn   = (wave / WM_WAVES) * 64;

  v8f acc[4][4];
  #pragma unroll
  for (int i = 0; i < 4; ++i)
    #pragma unroll
    for (int j = 0; j < 4; ++j)
      #pragma unroll
      for (int r = 0; r < 8; ++r) acc[i][j][r] = 0.0f;

  // ---- async staging plan: (BM+BN) rows x 64B per K-step = 1536 x 16B chunks,
  //      6 chunks per thread. LDS flat addr low 32 bits == DS byte offset.
  const unsigned ldsBase = (unsigned)(uintptr_t)&ldsbuf[0][0];
  const unsigned short* gp[6];
  unsigned lo[6];
  #pragma unroll
  for (int q = 0; q < 6; ++q) {
    const int c    = t + q * 256;
    const bool isA = (c < BM * 4);
    const int cc   = isA ? c : c - BM * 4;
    const int row  = cc >> 2;
    const int coff = (cc & 3) * 8;                     // halves within the row
    gp[q] = (isA ? Ab : Bb) + (long long)row * (isA ? lda : ldb) + coff;
    lo[q] = ldsBase + (unsigned)(((isA ? 0 : BM * LDSS) + row * LDSS + coff) * 2);
  }
  constexpr unsigned BUFB = (unsigned)(ROWS * LDSS * 2);

  // CDNA5 WMMA fragment addressing (ISA 7.12.2):
  //  A 16x32 bf16: lanes<16 hold K 0-7 (halves0-7) and 16-23 (halves8-15);
  //                lanes>=16 hold K 8-15 and 24-31.
  //  B 32x16 bf16: lane%16 = column n; same K split as A (lanes<16 K 0-7,16-23; lanes>=16 K 8-15,24-31).
  const int l16   = lane & 15;
  const int hi    = (lane & 16) ? 1 : 0;
  const int aKoff = hi ? 8 : 0;
  const int bKoff = hi ? 8 : 0;

  const int nk = K / KSTEP;

  // prologue: stage 0 into buffer 0
  #pragma unroll
  for (int q = 0; q < 6; ++q) async_load_b128(lo[q], gp[q]);

  for (int s = 0; s < nk; ++s) {
    wait_asynccnt0();          // own stage-s chunks are in LDS
    __syncthreads();           // everyone's stage-s data visible; stage s-1 reads done
    if (s + 1 < nk) {          // refill the other buffer under the WMMAs
      const int k0 = (s + 1) * KSTEP;
      const unsigned boff = (unsigned)((s + 1) & 1) * BUFB;
      #pragma unroll
      for (int q = 0; q < 6; ++q) async_load_b128(lo[q] + boff, gp[q] + k0);
    }

    const unsigned short* Abase = &ldsbuf[s & 1][0];
    const unsigned short* Bbase = &ldsbuf[s & 1][BM * LDSS];

    FragBF af[4];
    #pragma unroll
    for (int i = 0; i < 4; ++i) {
      const unsigned short* p = Abase + (wm + i * 16 + l16) * LDSS;
      af[i].u[0] = *(const uint4*)(p + aKoff);
      af[i].u[1] = *(const uint4*)(p + aKoff + 16);
    }
    #pragma unroll
    for (int j = 0; j < 4; ++j) {
      FragBF bfg;
      const unsigned short* p = Bbase + (wn + j * 16 + l16) * LDSS + bKoff;
      bfg.u[0] = *(const uint4*)(p);
      bfg.u[1] = *(const uint4*)(p + 16);
      #pragma unroll
      for (int i = 0; i < 4; ++i)
        acc[i][j] = __builtin_amdgcn_wmma_f32_16x16x32_bf16(
            false, af[i].v, false, bfg.v, (short)0, acc[i][j], false, false);
    }
  }

  // epilogue: C 16x16 f32 layout: VGPR r -> (M=r,N=lane) lanes<16; (M=8+r,N=lane-16)
  const int mTop = blockIdx.y * BM + wm + (hi ? 8 : 0);
  const int nTop = blockIdx.x * BN + wn + l16;
  #pragma unroll
  for (int i = 0; i < 4; ++i) {
    #pragma unroll
    for (int j = 0; j < 4; ++j) {
      const int n = nTop + j * 16;
      const float bv = bias ? bias[n] : 0.0f;
      #pragma unroll
      for (int r = 0; r < 8; ++r) {
        const int m = mTop + i * 16 + r;
        const float val = alpha * acc[i][j][r] + bv;
        if (omode == 0) {
          float* C = (float*)Cv + (long long)bz * sC;
          C[(long long)m * ldc + n] = val;
        } else if (omode == 1) {
          unsigned short* C = (unsigned short*)Cv + (long long)bz * sC;
          C[(long long)m * ldc + n] = f2bf(val);
        } else {
          unsigned short* C = (unsigned short*)Cv + (long long)bz * sC;
          C[(long long)n * ldc + m] = f2bf(val);
        }
      }
    }
  }
}

// ---------------------------------------------------------------------------
// Row softmax: f32 in -> bf16 probabilities out. One block (256 thr) per row.
// ---------------------------------------------------------------------------
__global__ __launch_bounds__(256) void softmax_rows_kernel(
    const float* __restrict__ in, unsigned short* __restrict__ out, int ncols)
{
  __shared__ float sred[256];
  const long long row = blockIdx.x;
  const float* x = in + row * ncols;
  unsigned short* y = out + row * ncols;
  const int t = threadIdx.x;

  float mx = -3.402823466e38f;
  for (int c = t; c < ncols; c += 256) mx = fmaxf(mx, x[c]);
  sred[t] = mx; __syncthreads();
  for (int s = 128; s > 0; s >>= 1) { if (t < s) sred[t] = fmaxf(sred[t], sred[t + s]); __syncthreads(); }
  mx = sred[0]; __syncthreads();

  float sum = 0.0f;
  for (int c = t; c < ncols; c += 256) sum += __expf(x[c] - mx);
  sred[t] = sum; __syncthreads();
  for (int s = 128; s > 0; s >>= 1) { if (t < s) sred[t] += sred[t + s]; __syncthreads(); }
  const float inv = 1.0f / sred[0];

  for (int c = t; c < ncols; c += 256) y[c] = f2bf(__expf(x[c] - mx) * inv);
}

// ---------------------------------------------------------------------------
// LayerNorm over last dim (E). One block per row; writes f32 + bf16 copies.
// ---------------------------------------------------------------------------
__global__ __launch_bounds__(256) void layernorm_rows_kernel(
    const float* __restrict__ in, const float* __restrict__ g, const float* __restrict__ b,
    float* __restrict__ outf, unsigned short* __restrict__ outbf, int E_)
{
  __shared__ float sred[256];
  const long long row = blockIdx.x;
  const float* x = in + row * E_;
  const int t = threadIdx.x;

  float s = 0.0f;
  for (int c = t; c < E_; c += 256) s += x[c];
  sred[t] = s; __syncthreads();
  for (int st = 128; st > 0; st >>= 1) { if (t < st) sred[t] += sred[t + st]; __syncthreads(); }
  const float mean = sred[0] / (float)E_; __syncthreads();

  float v = 0.0f;
  for (int c = t; c < E_; c += 256) { float d = x[c] - mean; v += d * d; }
  sred[t] = v; __syncthreads();
  for (int st = 128; st > 0; st >>= 1) { if (t < st) sred[t] += sred[t + st]; __syncthreads(); }
  const float rstd = rsqrtf(sred[0] / (float)E_ + 1e-5f);

  for (int c = t; c < E_; c += 256) {
    const float val = (x[c] - mean) * rstd * g[c] + b[c];
    outf[row * E_ + c]  = val;
    outbf[row * E_ + c] = f2bf(val);
  }
}

// ---------------------------------------------------------------------------
// GRU gate fuse: gi[M,3E], gh[M,3E], h[M,E] -> newh f32 + bf16
// ---------------------------------------------------------------------------
__global__ __launch_bounds__(256) void gru_fuse_kernel(
    const float* __restrict__ gi, const float* __restrict__ gh,
    const float* __restrict__ h,
    float* __restrict__ outf, unsigned short* __restrict__ outbf,
    int E_, long long total)
{
  const long long idx = (long long)blockIdx.x * 256 + threadIdx.x;
  if (idx >= total) return;
  const long long row = idx / E_;
  const int col = (int)(idx - row * E_);
  const long long b3 = row * (3LL * E_);
  const float ir  = gi[b3 + col],           hr = gh[b3 + col];
  const float iz  = gi[b3 + E_ + col],      hz = gh[b3 + E_ + col];
  const float in_ = gi[b3 + 2 * E_ + col],  hn = gh[b3 + 2 * E_ + col];
  const float r   = 1.0f / (1.0f + __expf(-(ir + hr)));
  const float zt  = 1.0f / (1.0f + __expf(-(iz + hz)));
  const float n   = tanhf(in_ + r * hn);
  const float hv  = h[idx];
  const float out = (1.0f - zt) * n + zt * hv;
  outf[idx]  = out;
  outbf[idx] = f2bf(out);
}

// ---------------------------------------------------------------------------
// f32 -> bf16 convert, and slots broadcast
// ---------------------------------------------------------------------------
__global__ __launch_bounds__(256) void cvt_bf16_kernel(
    const float* __restrict__ in, unsigned short* __restrict__ out, long long n)
{
  const long long i = (long long)blockIdx.x * 256 + threadIdx.x;
  if (i < n) out[i] = f2bf(in[i]);
}

__global__ __launch_bounds__(256) void bcast_slots_kernel(
    const float* __restrict__ slots, float* __restrict__ mem, long long per, long long total)
{
  const long long i = (long long)blockIdx.x * 256 + threadIdx.x;
  if (i < total) mem[i] = slots[i % per];
}

// ---------------------------------------------------------------------------
// Host orchestration
// ---------------------------------------------------------------------------
extern "C" void kernel_launch(void* const* d_in, const int* in_sizes, int n_in,
                              void* d_out, int out_size, void* d_ws, size_t ws_size,
                              hipStream_t stream) {
  (void)in_sizes; (void)n_in; (void)out_size; (void)ws_size;

  constexpr int cE = 768, cS = 128, cT = 3, cB = 64, cL = 512, cE3 = 3 * 768;
  constexpr long long BL  = (long long)cB * cL;         // 32768
  constexpr long long BS  = (long long)cB * cS;         // 8192
  constexpr long long BLE = BL * cE;
  constexpr long long BSE = BS * cE;
  constexpr long long BSL = (long long)cB * cS * cL;
  constexpr long long EE  = (long long)cE * cE;
  constexpr long long E3E = (long long)cE3 * cE;

  const float* z     = (const float*)d_in[0];
  const float* slots = (const float*)d_in[1];
  const float* ln_g  = (const float*)d_in[2];
  const float* ln_b  = (const float*)d_in[3];
  const float* w_wq = (const float*)d_in[4];  const float* w_bq = (const float*)d_in[5];
  const float* w_wk = (const float*)d_in[6];  const float* w_bk = (const float*)d_in[7];
  const float* w_wv = (const float*)d_in[8];  const float* w_bv = (const float*)d_in[9];
  const float* w_wo = (const float*)d_in[10]; const float* w_bo = (const float*)d_in[11];
  const float* r_wq = (const float*)d_in[12]; const float* r_bq = (const float*)d_in[13];
  const float* r_wk = (const float*)d_in[14]; const float* r_bk = (const float*)d_in[15];
  const float* r_wv = (const float*)d_in[16]; const float* r_bv = (const float*)d_in[17];
  const float* r_wo = (const float*)d_in[18]; const float* r_bo = (const float*)d_in[19];
  const float* gru_wih = (const float*)d_in[20]; const float* gru_bih = (const float*)d_in[21];
  const float* gru_whh = (const float*)d_in[22]; const float* gru_bhh = (const float*)d_in[23];

  // ---- workspace arena ----
  char* ws = (char*)d_ws;
  size_t off = 0;
  auto alloc = [&](size_t bytes) -> void* {
    off = (off + 255) & ~(size_t)255;
    void* p = ws + off;
    off += bytes;
    return p;
  };
  unsigned short* z_bf    = (unsigned short*)alloc(BLE * 2);
  unsigned short* Kz      = (unsigned short*)alloc(BLE * 2);   // [B][L][E]
  unsigned short* VzT     = (unsigned short*)alloc(BLE * 2);   // [B][E][L]
  unsigned short* Qr      = (unsigned short*)alloc(BLE * 2);   // read Q; reused as AVr
  unsigned short* wq_bf   = (unsigned short*)alloc(EE * 2);
  unsigned short* wk_bf   = (unsigned short*)alloc(EE * 2);
  unsigned short* wv_bf   = (unsigned short*)alloc(EE * 2);
  unsigned short* wo_bf   = (unsigned short*)alloc(EE * 2);
  unsigned short* rq_bf   = (unsigned short*)alloc(EE * 2);
  unsigned short* rk_bf   = (unsigned short*)alloc(EE * 2);
  unsigned short* rv_bf   = (unsigned short*)alloc(EE * 2);
  unsigned short* ro_bf   = (unsigned short*)alloc(EE * 2);
  unsigned short* wih_bf  = (unsigned short*)alloc(E3E * 2);
  unsigned short* whh_bf  = (unsigned short*)alloc(E3E * 2);
  float*          mem_f   = (float*)alloc(BSE * 4);
  float*          memn_f  = (float*)alloc(BSE * 4);
  unsigned short* memn_bf = (unsigned short*)alloc(BSE * 2);
  unsigned short* mem_bf  = (unsigned short*)alloc(BSE * 2);
  unsigned short* Qw      = (unsigned short*)alloc(BSE * 2);
  float*          scores  = (float*)alloc(BSL * 4);            // reused for read scores
  unsigned short* P_bf    = (unsigned short*)alloc(BSL * 2);   // reused for read probs
  unsigned short* AV_bf   = (unsigned short*)alloc(BSE * 2);
  unsigned short* upd_bf  = (unsigned short*)alloc(BSE * 2);
  float*          gi      = (float*)alloc(BS * cE3 * 4);
  float*          gh      = (float*)alloc(BS * cE3 * 4);
  unsigned short* Kr      = (unsigned short*)alloc(BSE * 2);
  unsigned short* VrT     = (unsigned short*)alloc(BSE * 2);   // [B][E][S]

  const float scale = 1.0f / sqrtf((float)cE);

  auto cvt = [&](const float* src, unsigned short* dst, long long n) {
    cvt_bf16_kernel<<<dim3((unsigned)((n + 255) / 256)), 256, 0, stream>>>(src, dst, n);
  };
  // cfg 0: 128x256 tile (N%256==0). cfg 1: 256x128 tile (M%256==0).
  auto gemm = [&](int cfg, const unsigned short* A, long long sA,
                  const unsigned short* Bw, long long sB,
                  void* C, long long sC, const float* bias,
                  int M, int N, int K, int lda, int ldb, int ldc,
                  float alpha, int omode, int batch) {
    if (cfg == 0) {
      dim3 g((unsigned)(N / 256), (unsigned)(M / 128), (unsigned)batch);
      gemm_bf16_kernel<128, 256><<<g, 256, 0, stream>>>(
          A, sA, Bw, sB, C, sC, bias, K, lda, ldb, ldc, alpha, omode);
    } else {
      dim3 g((unsigned)(N / 128), (unsigned)(M / 256), (unsigned)batch);
      gemm_bf16_kernel<256, 128><<<g, 256, 0, stream>>>(
          A, sA, Bw, sB, C, sC, bias, K, lda, ldb, ldc, alpha, omode);
    }
  };

  // ---- precompute (loop-invariant: z never changes across the recurrence) ----
  cvt(z, z_bf, BLE);
  cvt(w_wq, wq_bf, EE); cvt(w_wk, wk_bf, EE); cvt(w_wv, wv_bf, EE); cvt(w_wo, wo_bf, EE);
  cvt(r_wq, rq_bf, EE); cvt(r_wk, rk_bf, EE); cvt(r_wv, rv_bf, EE); cvt(r_wo, ro_bf, EE);
  cvt(gru_wih, wih_bf, E3E); cvt(gru_whh, whh_bf, E3E);
  bcast_slots_kernel<<<dim3((unsigned)((BSE + 255) / 256)), 256, 0, stream>>>(
      slots, mem_f, (long long)cS * cE, BSE);

  // K = z@Wk^T + bk                              -> bf16 [B*L, E]
  gemm(0, z_bf, 0, wk_bf, 0, Kz, 0, w_bk, (int)BL, cE, cE, cE, cE, cE, 1.0f, 1, 1);
  // V^T = (z@Wv^T + bv)^T per batch              -> bf16 [B][E][L]
  gemm(0, z_bf, (long long)cL * cE, wv_bf, 0, VzT, (long long)cE * cL, w_bv,
       cL, cE, cE, cE, cE, cL, 1.0f, 2, cB);
  // Q_read = z@r_wq^T + r_bq                     -> bf16 [B*L, E]
  gemm(0, z_bf, 0, rq_bf, 0, Qr, 0, r_bq, (int)BL, cE, cE, cE, cE, cE, 1.0f, 1, 1);

  // ---- recurrence: T = 3 ----
  for (int it = 0; it < cT; ++it) {
    layernorm_rows_kernel<<<dim3((unsigned)BS), 256, 0, stream>>>(
        mem_f, ln_g, ln_b, memn_f, memn_bf, cE);
    gemm(0, memn_bf, 0, wq_bf, 0, Qw, 0, w_bq, (int)BS, cE, cE, cE, cE, cE, 1.0f, 1, 1);
    // scores[b] = Qw_b [S,E] x Kz_b [L,E]^T * scale  -> f32 [B][S][L]
    gemm(0, Qw, (long long)cS * cE, Kz, (long long)cL * cE, scores, (long long)cS * cL,
         nullptr, cS, cL, cE, cE, cE, cL, scale, 0, cB);
    softmax_rows_kernel<<<dim3((unsigned)BS), 256, 0, stream>>>(scores, P_bf, cL);
    // AV[b] = P_b [S,L] x (V^T_b [E,L])^T            -> bf16 [B][S][E]
    gemm(0, P_bf, (long long)cS * cL, VzT, (long long)cE * cL, AV_bf, (long long)cS * cE,
         nullptr, cS, cE, cL, cL, cL, cE, 1.0f, 1, cB);
    gemm(0, AV_bf, 0, wo_bf, 0, upd_bf, 0, w_bo, (int)BS, cE, cE, cE, cE, cE, 1.0f, 1, 1);
    gemm(0, upd_bf,  0, wih_bf, 0, gi, 0, gru_bih, (int)BS, cE3, cE, cE, cE, cE3, 1.0f, 0, 1);
    gemm(0, memn_bf, 0, whh_bf, 0, gh, 0, gru_bhh, (int)BS, cE3, cE, cE, cE, cE3, 1.0f, 0, 1);
    gru_fuse_kernel<<<dim3((unsigned)((BSE + 255) / 256)), 256, 0, stream>>>(
        gi, gh, memn_f, mem_f, mem_bf, cE, BSE);
  }

  // ---- read attention ----
  gemm(0, mem_bf, 0, rk_bf, 0, Kr, 0, r_bk, (int)BS, cE, cE, cE, cE, cE, 1.0f, 1, 1);
  // Vr^T = (mem@r_wv^T + bv)^T per batch  -> [B][E][S]
  gemm(0, mem_bf, (long long)cS * cE, rv_bf, 0, VrT, (long long)cE * cS, r_bv,
       cS, cE, cE, cE, cE, cS, 1.0f, 2, cB);
  // scores_r[b] = Qr_b [L,E] x Kr_b [S,E]^T * scale -> f32 [B][L][S]  (256x128 tile)
  gemm(1, Qr, (long long)cL * cE, Kr, (long long)cS * cE, scores, (long long)cL * cS,
       nullptr, cL, cS, cE, cE, cE, cS, scale, 0, cB);
  softmax_rows_kernel<<<dim3((unsigned)BL), 256, 0, stream>>>(scores, P_bf, cS);
  // AVr[b] = Pr_b [L,S] x (Vr^T_b [E,S])^T -> bf16 [B][L][E]  (reuse Qr buffer)
  unsigned short* AVr = Qr;
  gemm(0, P_bf, (long long)cL * cS, VrT, (long long)cE * cS, AVr, (long long)cL * cE,
       nullptr, cL, cE, cS, cS, cS, cE, 1.0f, 1, cB);
  // out = AVr @ r_wo^T + bo  -> f32 d_out
  gemm(0, AVr, 0, ro_bf, 0, d_out, 0, r_bo, (int)BL, cE, cE, cE, cE, cE, 1.0f, 0, 1);
}
